// MultiHeadAttention_63118839382655
// MI455X (gfx1250) — matrix operand order repaired and checked by reading, not compiled, not run on hardware
//
#include <hip/hip_runtime.h>

#define D_MODEL   1024
#define NUM_HEADS 16
#define HEAD_DIM  64
#define BATCH     2
#define SEQ       2048
#define M_TOTAL   (BATCH * SEQ)   // 4096

typedef __attribute__((ext_vector_type(16))) __bf16 bf16x16;
typedef __attribute__((ext_vector_type(8)))  float  f32x8;
typedef __attribute__((ext_vector_type(4)))  float  f32x4;
typedef unsigned short u16;
typedef __attribute__((ext_vector_type(4)))  u16 u16x4;
typedef __attribute__((ext_vector_type(8)))  u16 u16x8;
typedef __attribute__((ext_vector_type(16))) u16 u16x16;

__device__ __forceinline__ u16 f2bf(float f) {
  union { float f; unsigned u; } c; c.f = f;
  unsigned u = c.u + 0x7FFFu + ((c.u >> 16) & 1u);   // round-to-nearest-even
  return (u16)(u >> 16);
}

// Assemble a 16-element bf16 fragment from two 16-byte LDS chunks.
__device__ __forceinline__ bf16x16 make16(const u16* lo, const u16* hi) {
  u16x8 a = *(const u16x8*)lo;
  u16x8 b = *(const u16x8*)hi;
  u16x16 t;
#pragma unroll
  for (int i = 0; i < 8; ++i) { t[i] = a[i]; t[i + 8] = b[i]; }
  return __builtin_bit_cast(bf16x16, t);
}

// ---------------------------------------------------------------------------
// GEMM: out[M,N] = (X[M,K] @ W[N,K]^T + bias) * oscale
// MODE 0: out is bf16 scattered to per-head [B, H, S, HEAD_DIM]
// MODE 1: out is fp32 row-major [M, N]
// WG = 256 threads (8 waves). WG tile 128x128; wave tile 64x32 (4x2 WMMA tiles).
// Software-pipelined: next k-tile staged in registers during WMMA compute.
// ---------------------------------------------------------------------------
template <int MODE>
__global__ __launch_bounds__(256)
void gemm_xwt(const float* __restrict__ X, const float* __restrict__ W,
              const float* __restrict__ bias, void* __restrict__ out,
              int M, int N, int K, float oscale) {
  __shared__ u16 As[128][40];   // 128x32 bf16 A tile (stride 80B, conflict-free)
  __shared__ u16 Bs[128][40];   // 128x32 bf16 B tile (rows of W)

  const int tid   = threadIdx.x;
  const int lane  = tid & 31;
  const int wave  = tid >> 5;
  const int waveM = wave >> 2;            // 0..1  -> 64-row strip
  const int waveN = wave & 3;             // 0..3  -> 32-col strip
  const int tileM = blockIdx.x * 128;
  const int tileN = blockIdx.y * 128;
  const int arow  = lane & 15;
  const int kgrp  = lane >> 4;

  f32x4 ra[4], rb[4];
  auto load_tile = [&](int k0) {
#pragma unroll
    for (int i = 0; i < 4; ++i) {
      int f = i * 256 + tid, row = f >> 3, c4 = (f & 7) * 4;
      ra[i] = *(const f32x4*)(X + (size_t)(tileM + row) * K + k0 + c4);
      rb[i] = *(const f32x4*)(W + (size_t)(tileN + row) * K + k0 + c4);
    }
  };
  auto store_tile = [&]() {
#pragma unroll
    for (int i = 0; i < 4; ++i) {
      int f = i * 256 + tid, row = f >> 3, c4 = (f & 7) * 4;
      u16x4 ap = { f2bf(ra[i].x), f2bf(ra[i].y), f2bf(ra[i].z), f2bf(ra[i].w) };
      *(u16x4*)&As[row][c4] = ap;
      u16x4 bp = { f2bf(rb[i].x), f2bf(rb[i].y), f2bf(rb[i].z), f2bf(rb[i].w) };
      *(u16x4*)&Bs[row][c4] = bp;
    }
  };

  f32x8 acc[4][2];
#pragma unroll
  for (int mt = 0; mt < 4; ++mt)
#pragma unroll
    for (int nt = 0; nt < 2; ++nt) acc[mt][nt] = (f32x8){};

  load_tile(0);
  store_tile();

  for (int k0 = 0; k0 < K; k0 += 32) {
    __syncthreads();                       // tile k0 visible in LDS
    const bool more = (k0 + 32) < K;
    if (more) {
      if (k0 + 64 < K) {                   // gfx1250 global_prefetch_b8
        __builtin_prefetch(X + (size_t)(tileM + (tid >> 1)) * K + k0 + 64 + (tid & 1) * 16, 0, 3);
        __builtin_prefetch(W + (size_t)(tileN + (tid >> 1)) * K + k0 + 64 + (tid & 1) * 16, 0, 3);
      }
      load_tile(k0 + 32);                  // global loads overlap WMMA below
    }

    bf16x16 bfr[2];
#pragma unroll
    for (int nt = 0; nt < 2; ++nt) {
      const u16* bp = &Bs[waveN * 32 + nt * 16 + arow][kgrp * 8];
      bfr[nt] = make16(bp, bp + 16);
    }
#pragma unroll
    for (int mt = 0; mt < 4; ++mt) {
      const u16* ap = &As[waveM * 64 + mt * 16 + arow][kgrp * 8];
      bf16x16 afr = make16(ap, ap + 16);   // K {0-7,16-23} / {8-15,24-31}
#pragma unroll
      for (int nt = 0; nt < 2; ++nt)
        acc[mt][nt] = __builtin_amdgcn_wmma_f32_16x16x32_bf16(
            false, afr, false, bfr[nt], (short)0, acc[mt][nt], false, false);
    }
    __syncthreads();                       // all waves done reading LDS
    if (more) store_tile();
  }

  // Epilogue: C-layout lane (r, group) -> row = r + group*8, col = lane&15
#pragma unroll
  for (int mt = 0; mt < 4; ++mt) {
#pragma unroll
    for (int nt = 0; nt < 2; ++nt) {
      int col = tileN + waveN * 32 + nt * 16 + (lane & 15);
      float bv = bias[col];
#pragma unroll
      for (int r = 0; r < 8; ++r) {
        int row = tileM + waveM * 64 + mt * 16 + kgrp * 8 + r;
        float v = (acc[mt][nt][r] + bv) * oscale;
        if (MODE == 0) {
          int b = row / SEQ, s = row % SEQ;
          int h = col / HEAD_DIM, d = col % HEAD_DIM;
          ((u16*)out)[(((size_t)b * NUM_HEADS + h) * SEQ + s) * HEAD_DIM + d] = f2bf(v);
        } else {
          ((float*)out)[(size_t)row * N + col] = v;
        }
      }
    }
  }
}

// ---------------------------------------------------------------------------
// V transpose: [B,H,S,64] bf16 -> [B,H,64,S] bf16, LDS-tiled, coalesced both ways.
// ---------------------------------------------------------------------------
__global__ __launch_bounds__(256)
void transpose_v(const u16* __restrict__ Vp, u16* __restrict__ Vt) {
  __shared__ u16 T[64][72];
  const int tid = threadIdx.x;
  const int bh  = blockIdx.y;
  const int s0  = blockIdx.x * 64;
  const u16* src = Vp + (size_t)bh * SEQ * HEAD_DIM;
  u16*       dst = Vt + (size_t)bh * HEAD_DIM * SEQ;
#pragma unroll
  for (int i = 0; i < 2; ++i) {
    int u = i * 256 + tid, row = u >> 3, c8 = (u & 7) * 8;
    *(u16x8*)&T[row][c8] = *(const u16x8*)(src + (size_t)(s0 + row) * HEAD_DIM + c8);
  }
  __syncthreads();
#pragma unroll
  for (int i = 0; i < 2; ++i) {
    int u = i * 256 + tid, d = u >> 3, c8 = (u & 7) * 8;
    u16x8 v;
#pragma unroll
    for (int e = 0; e < 8; ++e) v[e] = T[c8 + e][d];
    *(u16x8*)(dst + (size_t)d * SEQ + s0 + c8) = v;
  }
}

// ---------------------------------------------------------------------------
// Flash attention: one WG per (b*h, 128 q-rows). 8 waves, 16 q-rows each.
// K-tile = 64 keys. Q (pre-scaled by 1/sqrt(d)*log2e) in registers.
// K row-major and pre-transposed V staged to LDS, software-pipelined.
// ---------------------------------------------------------------------------
#define KT 64

__global__ __launch_bounds__(256)
void flash_attn(const u16* __restrict__ Qp, const u16* __restrict__ Kp,
                const u16* __restrict__ Vtg, float* __restrict__ ctx) {
  __shared__ u16 Ks[KT][72];          // [key][d]
  __shared__ u16 Vt[HEAD_DIM][72];    // [d][key] (already transposed in global)
  __shared__ u16 Ps[8][16][72];       // per-wave P tile (16 q x 64 k)

  const int tid  = threadIdx.x;
  const int lane = tid & 31;
  const int wave = tid >> 5;
  const int arow = lane & 15;
  const int kgrp = lane >> 4;
  const int bh   = blockIdx.y;
  const int qrow = blockIdx.x * 128 + wave * 16;

  const u16* Qh = Qp  + (size_t)bh * SEQ * HEAD_DIM;
  const u16* Kh = Kp  + (size_t)bh * SEQ * HEAD_DIM;
  const u16* Vh = Vtg + (size_t)bh * HEAD_DIM * SEQ;

  bf16x16 qfrag[2];
#pragma unroll
  for (int ks = 0; ks < 2; ++ks) {
    const u16* qp = Qh + (size_t)(qrow + arow) * HEAD_DIM + ks * 32 + kgrp * 8;
    qfrag[ks] = make16(qp, qp + 16);
  }

  u16x8 rk[2], rv[2];
  auto load_kv = [&](int jt) {
#pragma unroll
    for (int i = 0; i < 2; ++i) {
      int u = i * 256 + tid, row = u >> 3, c8 = (u & 7) * 8;
      rk[i] = *(const u16x8*)(Kh + (size_t)(jt * KT + row) * HEAD_DIM + c8);
      rv[i] = *(const u16x8*)(Vh + (size_t)row * SEQ + jt * KT + c8);
    }
  };
  auto store_kv = [&]() {
#pragma unroll
    for (int i = 0; i < 2; ++i) {
      int u = i * 256 + tid, row = u >> 3, c8 = (u & 7) * 8;
      *(u16x8*)&Ks[row][c8] = rk[i];
      *(u16x8*)&Vt[row][c8] = rv[i];
    }
  };

  float m[8], l[8];
  f32x8 O[4];
#pragma unroll
  for (int r = 0; r < 8; ++r) { m[r] = -3.0e38f; l[r] = 0.f; }
#pragma unroll
  for (int nt = 0; nt < 4; ++nt) O[nt] = (f32x8){};

  load_kv(0);
  store_kv();

  for (int jt = 0; jt < SEQ / KT; ++jt) {
    __syncthreads();                        // tile jt visible in LDS
    const bool more = (jt + 1) < SEQ / KT;
    if (more) {
      __builtin_prefetch(Kh + (size_t)((jt + 1) * KT + (tid >> 2)) * HEAD_DIM + (tid & 3) * 16, 0, 3);
      load_kv(jt + 1);                      // overlap with compute below
    }

    // Scores: Q (16x64) x K^T -> 16x64, 4 column tiles x 2 K-steps
    f32x8 sc[4];
#pragma unroll
    for (int nt = 0; nt < 4; ++nt) {
      f32x8 a = (f32x8){};
#pragma unroll
      for (int ks = 0; ks < 2; ++ks) {
        const u16* bp = &Ks[nt * 16 + arow][ks * 32 + kgrp * 8];
        bf16x16 bf = make16(bp, bp + 16);
        a = __builtin_amdgcn_wmma_f32_16x16x32_bf16(
            false, qfrag[ks], false, bf, (short)0, a, false, false);
      }
      sc[nt] = a;                           // already in log2 units (Q pre-scaled)
    }

    // Online softmax per q-row (row = r + kgrp*8 in C-layout)
    float alpha[8];
#pragma unroll
    for (int r = 0; r < 8; ++r) {
      float mx = sc[0][r];
#pragma unroll
      for (int nt = 1; nt < 4; ++nt) mx = fmaxf(mx, sc[nt][r]);
#pragma unroll
      for (int off = 1; off < 16; off <<= 1) mx = fmaxf(mx, __shfl_xor(mx, off, 32));
      float mn = fmaxf(m[r], mx);
      alpha[r] = exp2f(m[r] - mn);
      m[r] = mn;
      float rs = 0.f;
#pragma unroll
      for (int nt = 0; nt < 4; ++nt) {
        float p = exp2f(sc[nt][r] - mn);
        rs += p;
        Ps[wave][kgrp * 8 + r][nt * 16 + arow] = f2bf(p);   // C->A relayout
      }
#pragma unroll
      for (int off = 1; off < 16; off <<= 1) rs += __shfl_xor(rs, off, 32);
      l[r] = l[r] * alpha[r] + rs;
    }
#pragma unroll
    for (int nt = 0; nt < 4; ++nt)
#pragma unroll
      for (int r = 0; r < 8; ++r) O[nt][r] *= alpha[r];

    // O += P (16x64) x V (64x64): 2 K-steps x 4 d-tiles
#pragma unroll
    for (int ks = 0; ks < 2; ++ks) {
      const u16* ap = &Ps[wave][arow][ks * 32 + kgrp * 8];
      bf16x16 af = make16(ap, ap + 16);
#pragma unroll
      for (int nt = 0; nt < 4; ++nt) {
        const u16* bp = &Vt[nt * 16 + arow][ks * 32 + kgrp * 8];
        bf16x16 bf = make16(bp, bp + 16);
        O[nt] = __builtin_amdgcn_wmma_f32_16x16x32_bf16(
            false, af, false, bf, (short)0, O[nt], false, false);
      }
    }
    __syncthreads();                        // done reading LDS tile jt
    if (more) store_kv();
  }

  // Normalize; write ctx as [B, S, D_MODEL] fp32 (interleaved head layout)
  const int b = bh >> 4, h = bh & 15;
#pragma unroll
  for (int r = 0; r < 8; ++r) {
    float inv = 1.0f / l[r];
    int s = qrow + kgrp * 8 + r;
    float* orow = ctx + ((size_t)b * SEQ + s) * D_MODEL + h * HEAD_DIM;
#pragma unroll
    for (int nt = 0; nt < 4; ++nt) orow[nt * 16 + arow] = O[nt][r] * inv;
  }
}

// ---------------------------------------------------------------------------
extern "C" void kernel_launch(void* const* d_in, const int* in_sizes, int n_in,
                              void* d_out, int out_size, void* d_ws, size_t ws_size,
                              hipStream_t stream) {
  const float* query = (const float*)d_in[0];
  const float* key_  = (const float*)d_in[1];
  const float* value = (const float*)d_in[2];
  const float* Wq = (const float*)d_in[3];
  const float* bq = (const float*)d_in[4];
  const float* Wk = (const float*)d_in[5];
  const float* bk = (const float*)d_in[6];
  const float* Wv = (const float*)d_in[7];
  const float* bv = (const float*)d_in[8];
  const float* Wo = (const float*)d_in[9];
  const float* bo = (const float*)d_in[10];

  char* ws = (char*)d_ws;
  const size_t HSZ = (size_t)M_TOTAL * D_MODEL * sizeof(u16);   // 8 MB
  u16*   Qp  = (u16*)(ws);
  u16*   Kp  = (u16*)(ws + HSZ);
  u16*   Vp  = (u16*)(ws + 2 * HSZ);
  u16*   Vtg = (u16*)(ws + 3 * HSZ);
  float* ctx = (float*)(ws + 4 * HSZ);                          // 16 MB fp32

  const float CSC = 0.125f * 1.44269504089f;   // 1/sqrt(HEAD_DIM) * log2(e)

  dim3 gg(M_TOTAL / 128, D_MODEL / 128), gb(256);
  gemm_xwt<0><<<gg, gb, 0, stream>>>(query, Wq, bq, Qp, M_TOTAL, D_MODEL, D_MODEL, CSC);
  gemm_xwt<0><<<gg, gb, 0, stream>>>(key_,  Wk, bk, Kp, M_TOTAL, D_MODEL, D_MODEL, 1.0f);
  gemm_xwt<0><<<gg, gb, 0, stream>>>(value, Wv, bv, Vp, M_TOTAL, D_MODEL, D_MODEL, 1.0f);

  transpose_v<<<dim3(SEQ / 64, BATCH * NUM_HEADS), 256, 0, stream>>>(Vp, Vtg);

  flash_attn<<<dim3(SEQ / 128, BATCH * NUM_HEADS), 256, 0, stream>>>(Qp, Kp, Vtg, ctx);

  gemm_xwt<1><<<gg, gb, 0, stream>>>(ctx, Wo, bo, d_out, M_TOTAL, D_MODEL, D_MODEL, 1.0f);
}
